// GCNLayer_29712583753904
// MI455X (gfx1250) — matrix operand order repaired and checked by reading, not compiled, not run on hardware
//
#include <hip/hip_runtime.h>
#include <stdint.h>

#define NROWS 8192
#define D     128
#define KCHUNK 64
#define NCHUNK (NROWS / KCHUNK)

typedef __attribute__((ext_vector_type(16))) _Float16 v16h;
typedef __attribute__((ext_vector_type(8)))  _Float16 v8h;
typedef __attribute__((ext_vector_type(8)))  float    v8f;
typedef __attribute__((ext_vector_type(4)))  float    v4f;
typedef int v4i_vs __attribute__((vector_size(16)));   // builtin's pointee type

#if defined(__has_builtin)
#  if __has_builtin(__builtin_amdgcn_global_load_async_to_lds_b128) && \
      __has_builtin(__builtin_amdgcn_s_wait_asynccnt)
#    define USE_ASYNC_LDS 1
#  endif
#endif
#ifndef USE_ASYNC_LDS
#  define USE_ASYNC_LDS 0
#endif

// ---------------- Kernel 1: d[i] = rsqrt(sum_j adj[i][j]) ----------------
// Pure stream: non-temporal reads of adj so it never pollutes L2.
__global__ __launch_bounds__(256) void k_rowsum(const float* __restrict__ adj,
                                                float* __restrict__ dvec)
{
    __shared__ float red[256];
    const int row = blockIdx.x;
    const int t   = threadIdx.x;
    const v4f* rowp = (const v4f*)(adj + (size_t)row * NROWS);
    float s = 0.f;
#pragma unroll
    for (int i = 0; i < 8; ++i) {             // 8192 floats = 2048 float4
        v4f v = __builtin_nontemporal_load(rowp + t + 256 * i);
        s += v.x + v.y + v.z + v.w;
    }
    red[t] = s;
    __syncthreads();
    for (int off = 128; off > 0; off >>= 1) {
        if (t < off) red[t] += red[t + off];
        __syncthreads();
    }
    if (t == 0) {
        float rs = red[0];
        dvec[row] = (rs > 0.f) ? rsqrtf(rs) : 0.f;
    }
}

// ------- Kernel 2: yT[o][j] = d[j] * sum_k x[j,k]*W[o,k]  (f16, K-major) -------
__global__ __launch_bounds__(256) void k_xwT(const float* __restrict__ x,
                                             const float* __restrict__ W,
                                             const float* __restrict__ dvec,
                                             _Float16* __restrict__ yT)
{
    __shared__ float xs[32][D];               // 16 KB
    const int t  = threadIdx.x;
    const int j0 = blockIdx.x * 32;
    {
        const float4* src = (const float4*)(x + (size_t)j0 * D);
        float4* dst = (float4*)&xs[0][0];
#pragma unroll
        for (int i = 0; i < 4; ++i) dst[t + 256 * i] = src[t + 256 * i];
    }
    __syncthreads();

    const int o  = t & 127;                   // output feature
    const int jh = (t >> 7) * 16;             // 0 or 16: which 16 rows
    float acc[16];
#pragma unroll
    for (int j = 0; j < 16; ++j) acc[j] = 0.f;

    const float* wrow = W + (size_t)o * D;
    for (int k = 0; k < D; ++k) {
        float wk = wrow[k];
#pragma unroll
        for (int j = 0; j < 16; ++j)
            acc[j] += xs[jh + j][k] * wk;
    }
    _Float16* dst = yT + (size_t)o * NROWS + j0 + jh;
#pragma unroll
    for (int j = 0; j < 16; ++j)
        dst[j] = (_Float16)(acc[j] * dvec[j0 + jh + j]);
}

// ---- Kernel 3: out[i,:] = d[i] * (adj[i,:] @ y) + b  via v_wmma_f32_16x16x32_f16 ----
// Block: 32 rows x 128 cols. 512 threads = 16 waves; wave (mt,nt) owns one 16x16
// tile. Double-buffered LDS (one barrier per chunk); adj streamed NT from HBM
// exactly once; yT slice async-copied to LDS when the builtin is available.
__global__ __launch_bounds__(512) void k_spmm(const float* __restrict__ adj,
                                              const _Float16* __restrict__ yT,
                                              const float* __restrict__ dvec,
                                              const float* __restrict__ bias,
                                              float* __restrict__ out)
{
    __shared__ _Float16 aT[2][32][KCHUNK];    // adj tiles as f16:   2 x 4 KB
    __shared__ _Float16 bT[2][D][KCHUNK];     // yT tiles [col][k]:  2 x 16 KB

    const int t    = threadIdx.x;
    const int i0   = blockIdx.x * 32;
    const int wave = t >> 5;                  // 0..15
    const int lane = t & 31;
    const int mt   = wave >> 3;               // 0..1 : row sub-tile
    const int nt   = wave & 7;                // 0..7 : col tile

    // loader mappings
    const int lr  = t >> 4;                   // adj row within tile (0..31)
    const int lc4 = t & 15;                   // adj float4 col (0..15)

    auto load_chunk = [&](int k0, int buf) {
        // adj tile: 32 rows x 64 f32 = 512 float4 -> one per thread (NT), cvt f16
        v4f av = __builtin_nontemporal_load(
            (const v4f*)(adj + (size_t)(i0 + lr) * NROWS + k0) + lc4);
        _Float16* p = &aT[buf][lr][lc4 * 4];
        p[0] = (_Float16)av.x; p[1] = (_Float16)av.y;
        p[2] = (_Float16)av.z; p[3] = (_Float16)av.w;
        // yT tile: 128 cols x 64 halves = 1024 x 16B -> two b128 per thread
        int idx = t;
#pragma unroll
        for (int i = 0; i < 2; ++i, idx += 512) {
            const int c = idx >> 3;           // 0..127
            const int s = idx & 7;            // 8-half segment
            const _Float16* g = yT + (size_t)c * NROWS + k0 + s * 8;
#if USE_ASYNC_LDS
            __builtin_amdgcn_global_load_async_to_lds_b128(
                (v4i_vs*)g, (v4i_vs*)&bT[buf][c][s * 8], 0, 0);
#else
            *(v8h*)&bT[buf][c][s * 8] = *(const v8h*)g;
#endif
        }
    };

    v8f acc = {};

    load_chunk(0, 0);
#if USE_ASYNC_LDS
    __builtin_amdgcn_s_wait_asynccnt(0);
#endif
    __syncthreads();

    for (int it = 0; it < NCHUNK; ++it) {
        const int buf = it & 1;
        if (it + 1 < NCHUNK)
            load_chunk((it + 1) * KCHUNK, buf ^ 1);   // fill other buffer

#pragma unroll
        for (int kk = 0; kk < KCHUNK; kk += 32) {
            // A fragment (16x32 f16): lanes 0-15 -> K {0..7,16..23}; 16-31 -> K {8..15,24..31}
            const int r  = mt * 16 + (lane & 15);
            const int kb = (lane < 16) ? 0 : 8;
            v8h alo = *(const v8h*)&aT[buf][r][kk + kb];
            v8h ahi = *(const v8h*)&aT[buf][r][kk + kb + 16];
            // B fragment (32x16 f16): lanes 0-15 -> K {0..7,16..23}; 16-31 -> K {8..15,24..31}, col = lane%16
            const int c  = nt * 16 + (lane & 15);
            const int ko = kk + ((lane < 16) ? 0 : 8);
            v8h blo = *(const v8h*)&bT[buf][c][ko];
            v8h bhi = *(const v8h*)&bT[buf][c][ko + 16];
            v16h a, b;
#pragma unroll
            for (int i = 0; i < 8; ++i) {
                a[i] = alo[i]; a[8 + i] = ahi[i];
                b[i] = blo[i]; b[8 + i] = bhi[i];
            }
            acc = __builtin_amdgcn_wmma_f32_16x16x32_f16(
                      false, a, false, b, (short)0, acc, false, false);
        }

#if USE_ASYNC_LDS
        __builtin_amdgcn_s_wait_asynccnt(0);  // our LDS writes landed
#endif
        __syncthreads();                      // single barrier per chunk
    }

    // Epilogue: scale by d_i, add bias. C/D layout: VGPR v -> M = v + (lane<16?0:8)
    const int c     = nt * 16 + (lane & 15);
    const float bv  = bias[c];
    const int rbase = i0 + mt * 16 + ((lane < 16) ? 0 : 8);
#pragma unroll
    for (int v = 0; v < 8; ++v) {
        const int row = rbase + v;
        __builtin_nontemporal_store(acc[v] * dvec[row] + bv,
                                    out + (size_t)row * D + c);
    }
}

extern "C" void kernel_launch(void* const* d_in, const int* in_sizes, int n_in,
                              void* d_out, int out_size, void* d_ws, size_t ws_size,
                              hipStream_t stream) {
    (void)in_sizes; (void)n_in; (void)out_size; (void)ws_size;
    const float* x   = (const float*)d_in[0];   // [8192,128]
    const float* adj = (const float*)d_in[1];   // [8192,8192]
    const float* W   = (const float*)d_in[2];   // [128,128]
    const float* b   = (const float*)d_in[3];   // [128]
    float* out = (float*)d_out;                 // [8192,128]

    float*    dvec = (float*)d_ws;                                     // 32 KB
    _Float16* yT   = (_Float16*)((char*)d_ws + NROWS * sizeof(float)); // 2 MB

    k_rowsum<<<NROWS, 256, 0, stream>>>(adj, dvec);
    k_xwT  <<<NROWS / 32, 256, 0, stream>>>(x, W, dvec, yT);
    k_spmm <<<NROWS / 32, 512, 0, stream>>>(adj, yT, dvec, b, out);
}
